// HyperGRUCell_9543417332561
// MI455X (gfx1250) — compile-verified, matrix operand order checked by reading, not run on hardware
//
#include <hip/hip_runtime.h>
#include <cstdint>

// ---------------------------------------------------------------------------
// Types for CDNA5 WMMA
// ---------------------------------------------------------------------------
typedef __bf16  bf16;
typedef __bf16  v16bf __attribute__((ext_vector_type(16)));
typedef __bf16  bf16x4 __attribute__((ext_vector_type(4)));
typedef float   v8f   __attribute__((ext_vector_type(8)));

#define B_ROWS 16384
#define HDIM   512

// ---------------------------------------------------------------------------
// Conversion kernels
// ---------------------------------------------------------------------------
__global__ void cvt_weight_T(const float* __restrict__ Wr, const float* __restrict__ Ur,
                             const float* __restrict__ Wz, const float* __restrict__ Uz,
                             const float* __restrict__ W,  const float* __restrict__ U,
                             bf16* __restrict__ dst) {
    const float* srcs[6] = {Wr, Ur, Wz, Uz, W, U};
    const float* s = srcs[blockIdx.y];
    bf16* d = dst + (size_t)blockIdx.y * HDIM * HDIM;
    int idx = blockIdx.x * 256 + threadIdx.x;     // 0 .. 512*512-1
    int n = idx >> 9;
    int k = idx & (HDIM - 1);
    // transposed: d[n][k] = s[k][n]  -> B fragments become contiguous rows
    d[idx] = (bf16)s[(size_t)k * HDIM + n];
}

__global__ void cvt_bf16(const float* __restrict__ src, bf16* __restrict__ dst) {
    size_t i = ((size_t)blockIdx.x * blockDim.x + threadIdx.x) * 4;
    float4 v = *(const float4*)(src + i);
    bf16x4 o = {(bf16)v.x, (bf16)v.y, (bf16)v.z, (bf16)v.w};
    *(bf16x4*)(dst + i) = o;
}

// ---------------------------------------------------------------------------
// bf16 WMMA GEMM:  C[M=16384, N=512] = A[M,K=512] * Bt[N,K]^T   (f32 accum)
// Block: 256 threads = 8 waves. Tile: 128x128, K-step 32, double-buffered LDS
// fed by GLOBAL_LOAD_ASYNC_TO_LDS_B128 (ASYNCcnt-tracked, no VGPR round trip).
// Wave (wm in 0..1, wn in 0..3) owns a 64x32 sub-tile = 4x2 WMMA accumulators.
// ---------------------------------------------------------------------------
#define BM 128
#define BN 128
#define TK 32
#define NK (HDIM / TK)   // 16 K-steps
#define LDT 40           // padded LDS row stride in bf16 (80 bytes, 16B aligned)

__device__ __forceinline__ v16bf pack_frag(const bf16* base_lo, const bf16* base_hi) {
    union { float4 f[2]; v16bf v; } u;
    u.f[0] = *(const float4*)base_lo;
    u.f[1] = *(const float4*)base_hi;
    return u.v;
}

__device__ __forceinline__ void async_b128(const bf16* gsrc, bf16* lds_dst) {
    unsigned lds_addr = (unsigned)(uintptr_t)lds_dst;   // low 32 bits = LDS offset
    asm volatile("global_load_async_to_lds_b128 %0, %1, off"
                 :: "v"(lds_addr), "v"(gsrc) : "memory");
}

__global__ __launch_bounds__(256)
void gemm_bf16_wmma(const bf16* __restrict__ A, const bf16* __restrict__ Bt,
                    float* __restrict__ C) {
    __shared__ __attribute__((aligned(16))) bf16 As[2][BM * LDT];
    __shared__ __attribute__((aligned(16))) bf16 Bs[2][BN * LDT];

    const int tid  = threadIdx.x;
    const int lane = tid & 31;
    const int wave = tid >> 5;
    const int wm   = wave >> 2;    // 0..1
    const int wn   = wave & 3;     // 0..3
    const int g    = lane >> 4;    // K half-group (0/1) per 16-bit A layout
    const int mr   = lane & 15;    // M (or N) within 16

    const int block_row = blockIdx.y * BM;
    const int block_col = blockIdx.x * BN;

    // each thread owns two 8-bf16 chunks of the 128x32 tile (chunk = tid, tid+256)
    const int r0  = tid >> 2;                 // row 0..63
    const int cc0 = (tid & 3) << 3;           // col 0,8,16,24
    const int r1  = r0 + 64;                  // row 64..127
    const int cc1 = cc0;

    v8f acc[4][2] = {};

    // issue async copies of K-tile k0 into LDS buffer `buf` (4 x b128/thread)
    auto issue_tile = [&](int buf, int k0) {
        async_b128(A  + (size_t)(block_row + r0) * HDIM + k0 + cc0, &As[buf][r0 * LDT + cc0]);
        async_b128(A  + (size_t)(block_row + r1) * HDIM + k0 + cc1, &As[buf][r1 * LDT + cc1]);
        async_b128(Bt + (size_t)(block_col + r0) * HDIM + k0 + cc0, &Bs[buf][r0 * LDT + cc0]);
        async_b128(Bt + (size_t)(block_col + r1) * HDIM + k0 + cc1, &Bs[buf][r1 * LDT + cc1]);
    };

    issue_tile(0, 0);

    for (int kt = 0; kt < NK; ++kt) {
        const int buf = kt & 1;
        if (kt + 1 < NK) {
            issue_tile(buf ^ 1, (kt + 1) * TK);
            // current tile's 4 copies done; next tile's 4 may stay in flight
            asm volatile("s_wait_asynccnt 0x4" ::: "memory");
        } else {
            asm volatile("s_wait_asynccnt 0x0" ::: "memory");
        }
        __syncthreads();   // all waves' copies of this tile visible

        // ---- build fragments per 16-bit 16x32 A-matrix VGPR layout ----
        v16bf afrag[4], bfrag[2];
        #pragma unroll
        for (int mt = 0; mt < 4; ++mt) {
            int row = wm * 64 + mt * 16 + mr;
            afrag[mt] = pack_frag(&As[buf][row * LDT + g * 8],
                                  &As[buf][row * LDT + 16 + g * 8]);
        }
        #pragma unroll
        for (int nt = 0; nt < 2; ++nt) {
            int row = wn * 32 + nt * 16 + mr;
            bfrag[nt] = pack_frag(&Bs[buf][row * LDT + g * 8],
                                  &Bs[buf][row * LDT + 16 + g * 8]);
        }

        // ---- 8 WMMAs per K-step ----
        #pragma unroll
        for (int mt = 0; mt < 4; ++mt)
            #pragma unroll
            for (int nt = 0; nt < 2; ++nt)
                acc[mt][nt] = __builtin_amdgcn_wmma_f32_16x16x32_bf16(
                    false, afrag[mt], false, bfrag[nt],
                    (short)0, acc[mt][nt], false, false);

        __syncthreads();   // everyone done reading buf before it is re-filled
    }

    // ---- epilogue: C VGPR layout lane L, vgpr v -> M = v + 8*(L/16), N = L%16 ----
    #pragma unroll
    for (int mt = 0; mt < 4; ++mt) {
        #pragma unroll
        for (int nt = 0; nt < 2; ++nt) {
            int col = block_col + wn * 32 + nt * 16 + mr;
            #pragma unroll
            for (int v = 0; v < 8; ++v) {
                int row = block_row + wm * 64 + mt * 16 + v + 8 * g;
                C[(size_t)row * HDIM + col] = acc[mt][nt][v];
            }
        }
    }
}

// ---------------------------------------------------------------------------
// Row-wise hyperbolic elementwise stages. One wave (32 lanes) per row,
// 16 f32 per lane; reductions via wave32 butterfly shuffles.
// ---------------------------------------------------------------------------
__device__ __forceinline__ float wsum(float v) {
    v += __shfl_xor(v, 16, 32);
    v += __shfl_xor(v,  8, 32);
    v += __shfl_xor(v,  4, 32);
    v += __shfl_xor(v,  2, 32);
    v += __shfl_xor(v,  1, 32);
    return v;
}

__device__ __forceinline__ float artanh_clip(float z) {
    z = fminf(fmaxf(z, -1.f + 1e-6f), 1.f - 1e-6f);
    return atanhf(z);
}

// mx <- mobius_from_mx(x, mx)   (c = 1)
__device__ __forceinline__ void mobius_from_mx_ip(const float* x, float* mx) {
    float s1 = 0.f, s2 = 0.f;
    #pragma unroll
    for (int j = 0; j < 16; ++j) { s1 += x[j] * x[j]; s2 += mx[j] * mx[j]; }
    s1 = wsum(s1); s2 = wsum(s2);
    float xn   = sqrtf(fmaxf(s1, 1e-7f));
    float mxn  = sqrtf(fmaxf(s2, 1e-7f));
    float sc   = tanhf(mxn / xn * artanh_clip(xn)) / mxn;
    float sel  = (sqrtf(s2) > 1e-6f) ? sc : 0.f;
    #pragma unroll
    for (int j = 0; j < 16; ++j) mx[j] *= sel;
}

// x <- mobius_add(x, y)   (c = 1)
__device__ __forceinline__ void mobius_add_ip(float* x, const float* y) {
    float x2 = 0.f, y2 = 0.f, xy = 0.f;
    #pragma unroll
    for (int j = 0; j < 16; ++j) { x2 += x[j]*x[j]; y2 += y[j]*y[j]; xy += x[j]*y[j]; }
    x2 = wsum(x2); y2 = wsum(y2); xy = wsum(xy);
    float c1  = 1.f + 2.f * xy + y2;
    float c2  = 1.f - x2;
    float inv = 1.f / fmaxf(1.f + 2.f * xy + x2 * y2, 1e-7f);
    #pragma unroll
    for (int j = 0; j < 16; ++j) x[j] = (c1 * x[j] + c2 * y[j]) * inv;
}

// x <- logmap0(x)
__device__ __forceinline__ void logmap0_ip(float* x) {
    float s = 0.f;
    #pragma unroll
    for (int j = 0; j < 16; ++j) s += x[j] * x[j];
    s = wsum(s);
    float xn = sqrtf(fmaxf(s, 1e-7f));
    float sc = artanh_clip(xn) / xn;
    #pragma unroll
    for (int j = 0; j < 16; ++j) x[j] *= sc;
}

// gate = sigmoid(logmap0(mobius_add(hyp_linear(prev,*), hyp_linear(inp,*))))
// mode 0: out_pr = bf16(prev * gate)   mode 1: out_z = gate (f32)
__global__ __launch_bounds__(256)
void ew_gate(const float* __restrict__ prev, const float* __restrict__ inp,
             const float* __restrict__ Mw, const float* __restrict__ Mu,
             const float* __restrict__ bw, const float* __restrict__ bu,
             bf16* __restrict__ out_pr, float* __restrict__ out_z, int mode) {
    const int lane = threadIdx.x & 31;
    const size_t base = ((size_t)blockIdx.x * 8 + (threadIdx.x >> 5)) * HDIM;

    float p[16], a1[16], a2[16], q[16], tmp[16];
    #pragma unroll
    for (int j = 0; j < 16; ++j) {
        int c = lane + 32 * j;
        p[j]  = prev[base + c];
        a1[j] = Mw[base + c];
        q[j]  = inp[base + c];
        a2[j] = Mu[base + c];
    }
    mobius_from_mx_ip(p, a1);
    #pragma unroll
    for (int j = 0; j < 16; ++j) tmp[j] = bw[lane + 32 * j];
    mobius_add_ip(a1, tmp);

    mobius_from_mx_ip(q, a2);
    #pragma unroll
    for (int j = 0; j < 16; ++j) tmp[j] = bu[lane + 32 * j];
    mobius_add_ip(a2, tmp);

    mobius_add_ip(a1, a2);
    logmap0_ip(a1);

    #pragma unroll
    for (int j = 0; j < 16; ++j) {
        int c = lane + 32 * j;
        float r = 1.f / (1.f + expf(-a1[j]));
        if (mode == 0) out_pr[base + c] = (bf16)(p[j] * r);
        else           out_z[base + c]  = r;
    }
}

// out = mobius_add(mobius_from_mx(x, Mx), b)    (= hyp_linear given Mx = x@W)
__global__ __launch_bounds__(256)
void ew_hyplin(const float* __restrict__ x, const float* __restrict__ Mx,
               const float* __restrict__ b, float* __restrict__ out) {
    const int lane = threadIdx.x & 31;
    const size_t base = ((size_t)blockIdx.x * 8 + (threadIdx.x >> 5)) * HDIM;
    float xv[16], a[16], tmp[16];
    #pragma unroll
    for (int j = 0; j < 16; ++j) {
        int c = lane + 32 * j;
        xv[j] = x[base + c];
        a[j]  = Mx[base + c];
    }
    mobius_from_mx_ip(xv, a);
    #pragma unroll
    for (int j = 0; j < 16; ++j) tmp[j] = b[lane + 32 * j];
    mobius_add_ip(a, tmp);
    #pragma unroll
    for (int j = 0; j < 16; ++j) out[base + lane + 32 * j] = a[j];
}

// final: ht = mobius_add(prev, mobius_from_mx(res1, res1*zt)) with
// res1 = mobius_add(-prev, mobius_add(mobius_add(mfm(prev,M5),bW), A6))
__global__ __launch_bounds__(256)
void ew_final(const float* __restrict__ prev, const float* __restrict__ M5,
              const float* __restrict__ bW, const float* __restrict__ a6,
              const float* __restrict__ zt, float* __restrict__ out) {
    const int lane = threadIdx.x & 31;
    const size_t base = ((size_t)blockIdx.x * 8 + (threadIdx.x >> 5)) * HDIM;
    float p[16], h[16], tmp[16], negp[16], rz[16];
    #pragma unroll
    for (int j = 0; j < 16; ++j) {
        int c = lane + 32 * j;
        p[j] = prev[base + c];
        h[j] = M5[base + c];
    }
    mobius_from_mx_ip(p, h);
    #pragma unroll
    for (int j = 0; j < 16; ++j) tmp[j] = bW[lane + 32 * j];
    mobius_add_ip(h, tmp);                       // ht_new_h
    #pragma unroll
    for (int j = 0; j < 16; ++j) tmp[j] = a6[base + lane + 32 * j];
    mobius_add_ip(h, tmp);                       // ht_new
    #pragma unroll
    for (int j = 0; j < 16; ++j) negp[j] = -p[j];
    mobius_add_ip(negp, h);                      // res1
    #pragma unroll
    for (int j = 0; j < 16; ++j) rz[j] = negp[j] * zt[base + lane + 32 * j];
    mobius_from_mx_ip(negp, rz);                 // res2
    mobius_add_ip(p, rz);                        // ht
    #pragma unroll
    for (int j = 0; j < 16; ++j) out[base + lane + 32 * j] = p[j];
}

// ---------------------------------------------------------------------------
// Host launcher
// ---------------------------------------------------------------------------
extern "C" void kernel_launch(void* const* d_in, const int* in_sizes, int n_in,
                              void* d_out, int out_size, void* d_ws, size_t ws_size,
                              hipStream_t stream) {
    const float* inp  = (const float*)d_in[0];
    const float* prev = (const float*)d_in[1];
    const float* Wr  = (const float*)d_in[2];   const float* bWr = (const float*)d_in[3];
    const float* Ur  = (const float*)d_in[4];   const float* bUr = (const float*)d_in[5];
    const float* Wz  = (const float*)d_in[6];   const float* bWz = (const float*)d_in[7];
    const float* Uz  = (const float*)d_in[8];   const float* bUz = (const float*)d_in[9];
    const float* W   = (const float*)d_in[10];  const float* bW  = (const float*)d_in[11];
    const float* U   = (const float*)d_in[12];  const float* bU  = (const float*)d_in[13];
    float* out = (float*)d_out;

    char* ws = (char*)d_ws;
    size_t off = 0;
    auto alloc = [&](size_t bytes) -> char* {
        char* p = ws + off;
        off += (bytes + 255) & ~(size_t)255;
        return p;
    };
    const size_t BH  = (size_t)B_ROWS * HDIM;
    const size_t WEL = (size_t)HDIM * HDIM;

    bf16*  wt      = (bf16*)alloc(6 * WEL * sizeof(bf16));   // 6 transposed bf16 weights
    bf16*  prev_bf = (bf16*)alloc(BH * sizeof(bf16));
    bf16*  inp_bf  = (bf16*)alloc(BH * sizeof(bf16));
    float* M0      = (float*)alloc(BH * sizeof(float));
    float* M1      = (float*)alloc(BH * sizeof(float));
    bf16*  pr_bf   = (bf16*)alloc(BH * sizeof(bf16));        // bf16(prev * rt)
    float* ztb     = (float*)alloc(BH * sizeof(float));
    float* a6      = (float*)alloc(BH * sizeof(float));

    // 1. weight convert + transpose (6 x 512x512)
    cvt_weight_T<<<dim3(WEL / 256, 6), 256, 0, stream>>>(Wr, Ur, Wz, Uz, W, U, wt);
    // 2. activation convert
    cvt_bf16<<<BH / (256 * 4), 256, 0, stream>>>(prev, prev_bf);
    cvt_bf16<<<BH / (256 * 4), 256, 0, stream>>>(inp,  inp_bf);

    const dim3 gg(HDIM / BN, B_ROWS / BM);   // (4, 128)
    const int  ewg = B_ROWS / 8;             // 2048 blocks, 8 rows each

    // 3. reset gate GEMMs + gate fuse -> pr_bf = bf16(prev * rt)
    gemm_bf16_wmma<<<gg, 256, 0, stream>>>(prev_bf, wt + 0 * WEL, M0);
    gemm_bf16_wmma<<<gg, 256, 0, stream>>>(inp_bf,  wt + 1 * WEL, M1);
    ew_gate<<<ewg, 256, 0, stream>>>(prev, inp, M0, M1, bWr, bUr, pr_bf, nullptr, 0);

    // 4. update gate GEMMs + gate fuse -> ztb
    gemm_bf16_wmma<<<gg, 256, 0, stream>>>(prev_bf, wt + 2 * WEL, M0);
    gemm_bf16_wmma<<<gg, 256, 0, stream>>>(inp_bf,  wt + 3 * WEL, M1);
    ew_gate<<<ewg, 256, 0, stream>>>(prev, inp, M0, M1, bWz, bUz, nullptr, ztb, 1);

    // 5. A6 = hyp_linear(inp, U, bU)
    gemm_bf16_wmma<<<gg, 256, 0, stream>>>(inp_bf, wt + 5 * WEL, M0);
    ew_hyplin<<<ewg, 256, 0, stream>>>(inp, M0, bU, a6);

    // 6. M5 = (prev*rt) @ W, then final hyperbolic combine -> out
    gemm_bf16_wmma<<<gg, 256, 0, stream>>>(pr_bf, wt + 4 * WEL, M0);
    ew_final<<<ewg, 256, 0, stream>>>(prev, M0, bW, a6, ztb, out);

    (void)in_sizes; (void)n_in; (void)out_size; (void)ws_size;
}
